// PredRNN_80418967651021
// MI455X (gfx1250) — matrix operand order repaired and checked by reading, not compiled, not run on hardware
//
#include <hip/hip_runtime.h>

// ---------------------------------------------------------------------------
// PredRNN (SpatioTemporalLSTM, 3x64 hidden, 32x32, B=8, T=12) forward on
// gfx1250.  All convolutions run as implicit-im2col GEMMs on the WMMA matrix
// pipe (v_wmma_f32_16x16x32_f16, f16 in / f32 accumulate).  Weight tiles are
// DMA'd into LDS by the Tensor Data Mover (tensor_load_to_lds + TENSORcnt),
// overlapped with the per-thread im2col gather of the activation tile.
// Elementwise gate math and LayerNorm stay fp32.  Working set (~55 MB) lives
// in the 192 MB L2, so the kernel set is compute-bound.
// ---------------------------------------------------------------------------

typedef __attribute__((ext_vector_type(16))) _Float16 v16h;
typedef __attribute__((ext_vector_type(8)))  _Float16 v8h;
typedef __attribute__((ext_vector_type(8)))  float    v8f;
typedef __attribute__((ext_vector_type(4)))  unsigned u32x4;
typedef __attribute__((ext_vector_type(8)))  unsigned u32x8;

#define BATCH   8
#define TT      12
#define STEPS   11          // T-1
#define IN_NUM  6
#define FC      16
#define HC      64
#define HWSZ    1024        // 32*32
#define NTOT    (BATCH * HWSZ)   // 8192 GEMM-N
#define OUT_ELEMS (BATCH * STEPS * HWSZ * FC)   // 1441792

__device__ __forceinline__ float sigf(float x) { return 1.f / (1.f + expf(-x)); }

// ---------------------------------------------------------------------------
// Weight pack: fp32 [Cout,Cin,k,k] -> f16 [Cout,Kpad] row-major, zero-padded.
// OIHW flattening == our k = c*k*k + r*k + s ordering, so it is a plain copy.
// ---------------------------------------------------------------------------
__global__ void pack_w_kernel(const float* __restrict__ w, _Float16* __restrict__ o,
                              int Cout, int K, int Kpad)
{
    int e = blockIdx.x * blockDim.x + threadIdx.x;
    if (e >= Cout * Kpad) return;
    int m = e / Kpad;
    int k = e - m * Kpad;
    o[e] = (k < K) ? (_Float16)w[(size_t)m * K + k] : (_Float16)0.f;
}

// ---------------------------------------------------------------------------
// Implicit-im2col GEMM conv:  C[b,M,hw] = sum_k A[M,k] * im2col(X)[k, n]
//   A : f16 [Mtot, Kpad]   (packed weights, Kpad % 32 == 0)
//   X : f16 [B, Cin, 32, 32]
//   C : f32 [B, Mtot, 32, 32]
// Block = 256 threads = 8 wave32s.  WG tile 128(M) x 64(N); each wave owns a
// 32x32 tile -> 4 accumulators of v_wmma_f32_16x16x32_f16.
// The 128x32 f16 A-tile is loaded by the Tensor Data Mover (async DMA to LDS,
// D# built in SGPRs, issued once per workgroup by wave 0, TENSORcnt-tracked),
// overlapping with the VALU/VMEM im2col gather of the B-tile.
// ---------------------------------------------------------------------------
#define LDA 32   // sA pitch (halfs): matches TDM tile rows, 16B aligned
#define LDB 40   // sB pitch (halfs): 16B aligned, bank-conflict-mitigating

__launch_bounds__(256)
__global__ void conv_gemm_wmma(const _Float16* __restrict__ A,
                               const _Float16* __restrict__ X,
                               float* __restrict__ C,
                               int Mtot, int Cin, int Kpad, int KS)
{
    __shared__ _Float16 sA[128 * LDA];
    __shared__ _Float16 sB[64 * LDB];

    const int tid  = threadIdx.x;
    const int lane = tid & 31;
    const int wid  = tid >> 5;

    const int m0   = blockIdx.y * 128;
    const int n0   = blockIdx.x * 64;          // 64 | 1024 => one batch per tile
    const int K    = Cin * KS * KS;
    const int PADK = KS >> 1;
    const int b    = n0 >> 10;
    const int hw0  = n0 & 1023;

    const int waveM = (wid >> 1) * 32;         // 0,32,64,96
    const int waveN = (wid & 1) * 32;          // 0,32

    v8f acc[2][2];
#pragma unroll
    for (int mi = 0; mi < 2; ++mi)
#pragma unroll
        for (int ni = 0; ni < 2; ++ni)
#pragma unroll
            for (int v = 0; v < 8; ++v) acc[mi][ni][v] = 0.f;

    const int bn  = tid & 63;                  // B-loader: local n
    const int bkc = tid >> 6;                  // B-loader: k-chunk 0..3
    const int by  = (hw0 + bn) >> 5;
    const int bx  = (hw0 + bn) & 31;

    // ---- TDM descriptor group 1 (static per kernel, wave-uniform) ----
    // data_size=2B; tensor = [Mtot rows, Kpad cols] row-major;
    // tile = [tdim1 rows, 32 cols]; rows beyond Mtot zero-fill via OOB rules.
    const int tdim1 = (Mtot - m0 < 128) ? (Mtot - m0) : 128;
    u32x8 g1;
    g1[0] = 0x00010000u;                                   // wg_mask=0, data_size=1 (2B)
    g1[1] = ((unsigned)Kpad & 0xFFFFu) << 16;              // tensor_dim0[15:0]
    g1[2] = ((unsigned)Kpad >> 16) | (((unsigned)Mtot & 0xFFFFu) << 16); // dim0 hi, dim1 lo
    g1[3] = ((unsigned)Mtot >> 16) | (32u << 16);          // dim1 hi, tile_dim0=32
    g1[4] = (unsigned)tdim1;                               // tile_dim1, tile_dim2=0
    g1[5] = (unsigned)Kpad;                                // tensor_dim0_stride[31:0]
    g1[6] = 0u;                                            // stride hi, dim1_stride lo
    g1[7] = 0u;
    // LDS byte address of sA (flat->LDS addressing truncates to addr[31:0]).
    const unsigned ldsA = (unsigned)(unsigned long long)(const void*)sA;

    for (int k0 = 0; k0 < Kpad; k0 += 32) {
        // ---- issue TDM: A slab [m0..m0+127, k0..k0+31] -> sA (wave 0 only) ----
        if (wid == 0) {
            unsigned long long ga =
                (unsigned long long)(const void*)(A + (size_t)m0 * Kpad + k0);
            u32x4 g0;
            g0[0] = 1u;                                     // count=1, user mode
            g0[1] = ldsA;                                   // lds_addr
            g0[2] = (unsigned)ga;                           // global_addr[31:0]
            g0[3] = ((unsigned)(ga >> 32) & 0x01FFFFFFu)    // global_addr[56:32]
                    | (2u << 30);                           // type=2 ("image")
            asm volatile("tensor_load_to_lds %0, %1" :: "s"(g0), "s"(g1) : "memory");
        }
        if (tid == 0 && (k0 + 32) < Kpad)
            __builtin_prefetch(A + (size_t)m0 * Kpad + k0 + 32, 0, 1);

        // ---- stage B tile (implicit im2col gather), overlaps the DMA ----
#pragma unroll
        for (int j = 0; j < 8; ++j) {
            int kidx = k0 + bkc * 8 + j;
            _Float16 v = (_Float16)0.f;
            if (kidx < K) {
                int kk2 = KS * KS;
                int c   = kidx / kk2;
                int rs  = kidx - c * kk2;
                int r   = rs / KS;
                int s   = rs - r * KS;
                int iy  = by + r - PADK;
                int ix  = bx + s - PADK;
                if ((unsigned)iy < 32u && (unsigned)ix < 32u)
                    v = X[(((size_t)b * Cin + c) << 10) + (iy << 5) + ix];
            }
            sB[bn * LDB + bkc * 8 + j] = v;
        }
        if (wid == 0)
            __builtin_amdgcn_s_wait_tensorcnt(0);   // drain TDM before barrier
        __syncthreads();

        // ---- WMMA over the staged 32-K block ----
#pragma unroll
        for (int mi = 0; mi < 2; ++mi) {
            // A fragment (16x32 f16): lanes 0-15 hold K 0-7 / 16-23 of row M,
            // lanes 16-31 hold K 8-15 / 24-31 (ISA 7.12.2).
            int row = waveM + mi * 16 + (lane & 15);
            int kLo = (lane >> 4) * 8;
            v8h alo = *(const v8h*)(sA + row * LDA + kLo);
            v8h ahi = *(const v8h*)(sA + row * LDA + kLo + 16);
            v16h af = __builtin_shufflevector(alo, ahi,
                        0,1,2,3,4,5,6,7,8,9,10,11,12,13,14,15);
#pragma unroll
            for (int ni = 0; ni < 2; ++ni) {
                // B fragment (32x16 f16): lanes 0-15 hold K 0-7 / 16-23 of col N,
                // lanes 16-31 hold K 8-15 / 24-31 (same order as A).
                int col = waveN + ni * 16 + (lane & 15);
                v8h blo = *(const v8h*)(sB + col * LDB + kLo);
                v8h bhi = *(const v8h*)(sB + col * LDB + kLo + 16);
                v16h bf = __builtin_shufflevector(blo, bhi,
                            0,1,2,3,4,5,6,7,8,9,10,11,12,13,14,15);
                acc[mi][ni] = __builtin_amdgcn_wmma_f32_16x16x32_f16(
                    false, af, false, bf, (short)0, acc[mi][ni], false, false);
            }
        }
        __syncthreads();
    }

    // ---- store C (f32, NCHW) per documented 16x16 f32 C/D layout ----
    const int chBase = b * Mtot;
#pragma unroll
    for (int mi = 0; mi < 2; ++mi)
#pragma unroll
        for (int ni = 0; ni < 2; ++ni) {
            int Mb = m0 + waveM + mi * 16 + ((lane >> 4) << 3);
            int hw = hw0 + waveN + ni * 16 + (lane & 15);
#pragma unroll
            for (int v = 0; v < 8; ++v) {
                int M = Mb + v;
                if (M < Mtot)
                    C[(((size_t)(chBase + M)) << 10) + hw] = acc[mi][ni][v];
            }
        }
}

// ---------------------------------------------------------------------------
// LayerNorm over (C,H,W) per sample, in place.  grid = BATCH blocks.
// ---------------------------------------------------------------------------
__launch_bounds__(256)
__global__ void layernorm_kernel(float* __restrict__ x, int n)
{
    float* p = x + (size_t)blockIdx.x * n;
    __shared__ float rs[256], rq[256];
    float s = 0.f, q = 0.f;
    for (int i = threadIdx.x; i < n; i += 256) { float v = p[i]; s += v; q += v * v; }
    rs[threadIdx.x] = s; rq[threadIdx.x] = q;
    __syncthreads();
    for (int o = 128; o > 0; o >>= 1) {
        if (threadIdx.x < o) { rs[threadIdx.x] += rs[threadIdx.x + o];
                               rq[threadIdx.x] += rq[threadIdx.x + o]; }
        __syncthreads();
    }
    float inv = 1.f / (float)n;
    float mu  = rs[0] * inv;
    float var = rq[0] * inv - mu * mu;
    float rr  = rsqrtf(var + 1e-5f);
    for (int i = threadIdx.x; i < n; i += 256) p[i] = (p[i] - mu) * rr;
}

// ---------------------------------------------------------------------------
// Fused ST-LSTM gate math (per cell).  524288 threads = B * H * HW.
// Writes c,m states (fp32, in place), mem=[c;m] f16, m f16, o_pre = ox+oh.
// ---------------------------------------------------------------------------
__launch_bounds__(256)
__global__ void gates_kernel(const float* __restrict__ xcc,
                             const float* __restrict__ hcc,
                             const float* __restrict__ mcc,
                             float* __restrict__ c_st, float* __restrict__ m_st,
                             float* __restrict__ o_pre,
                             _Float16* __restrict__ mem16,
                             _Float16* __restrict__ m16)
{
    const int CS = HC * HWSZ;                   // 65536
    int e = blockIdx.x * blockDim.x + threadIdx.x;   // < 8*65536
    int b   = e >> 16;
    int rem = e & 65535;
    const float* xb = xcc + (size_t)b * 7 * CS;
    const float* hb = hcc + (size_t)b * 4 * CS;
    const float* mb = mcc + (size_t)b * 3 * CS;

    float ix  = xb[rem],          fx  = xb[rem + CS],     gx  = xb[rem + 2 * CS];
    float ixp = xb[rem + 3 * CS], fxp = xb[rem + 4 * CS], gxp = xb[rem + 5 * CS];
    float ox  = xb[rem + 6 * CS];
    float ih = hb[rem], fh = hb[rem + CS], gh = hb[rem + 2 * CS], oh = hb[rem + 3 * CS];
    float im = mb[rem], fm = mb[rem + CS], gm = mb[rem + 2 * CS];

    float c = c_st[e], m = m_st[e];
    float it = sigf(ix + ih);
    float ft = sigf(fx + fh + 1.0f);
    float gt = tanhf(gx + gh);
    float cn = ft * c + it * gt;
    float ip = sigf(ixp + im);
    float fp = sigf(fxp + fm + 1.0f);
    float gp = tanhf(gxp + gm);
    float mn = fp * m + ip * gp;

    c_st[e] = cn;
    m_st[e] = mn;
    o_pre[e] = ox + oh;
    size_t mbase = ((size_t)b * 2 * HC) << 10;   // mem is [B,128,32,32]
    mem16[mbase + rem]          = (_Float16)cn;
    mem16[mbase + rem + CS]     = (_Float16)mn;
    m16[e] = (_Float16)mn;
}

// o_t = sigmoid(o_pre + LN(conv_o(mem))), h = o_t * tanh(conv1x1(mem))
__launch_bounds__(256)
__global__ void hout_kernel(const float* __restrict__ o_pre,
                            const float* __restrict__ occ,
                            const float* __restrict__ t11,
                            _Float16* __restrict__ h16)
{
    int e = blockIdx.x * blockDim.x + threadIdx.x;
    float o = sigf(o_pre[e] + occ[e]);
    h16[e] = (_Float16)(o * tanhf(t11[e]));
}

// net = mask*frame + (1-mask)*x_gen   (frames/mask are NHWC fp32; net f16 NCHW)
__launch_bounds__(256)
__global__ void blend_kernel(const float* __restrict__ frames,
                             const float* __restrict__ mask_true,
                             const float* __restrict__ xgen,
                             _Float16* __restrict__ net16, int t)
{
    int e = blockIdx.x * blockDim.x + threadIdx.x;   // < 8*16*1024
    int b  = e >> 14;
    int ch = (e >> 10) & 15;
    int hw = e & 1023;
    float fr = frames[((((size_t)b * TT + t) << 10) + hw) * FC + ch];
    float mk = (t < IN_NUM) ? 1.f
             : mask_true[((((size_t)b * (STEPS - IN_NUM) + (t - IN_NUM)) << 10) + hw) * FC + ch];
    net16[e] = (_Float16)(mk * fr + (1.f - mk) * xgen[e]);
}

// d_out next_frames (NHWC, [B,11,32,32,16]) + MSE loss accumulation.
__launch_bounds__(256)
__global__ void emit_kernel(const float* __restrict__ xgen,
                            const float* __restrict__ frames,
                            float* __restrict__ out, float* __restrict__ loss, int t)
{
    int e = blockIdx.x * blockDim.x + threadIdx.x;   // < 8*16*1024 (NCHW xgen)
    int b  = e >> 14;
    int ch = (e >> 10) & 15;
    int hw = e & 1023;
    float v = xgen[e];
    out[((((size_t)b * STEPS + t) << 10) + hw) * FC + ch] = v;
    float g = frames[((((size_t)b * TT + (t + 1)) << 10) + hw) * FC + ch];
    float d = v - g;
    atomicAdd(loss, d * d * (1.0f / (float)OUT_ELEMS));
}

// ---------------------------------------------------------------------------
// Host orchestration
// ---------------------------------------------------------------------------
static inline void launch_gemm(hipStream_t st, const _Float16* A, const _Float16* X,
                               float* C, int Mtot, int Cin, int Kpad, int KS)
{
    dim3 grid(NTOT / 64, (Mtot + 127) / 128);
    conv_gemm_wmma<<<grid, 256, 0, st>>>(A, X, C, Mtot, Cin, Kpad, KS);
}

extern "C" void kernel_launch(void* const* d_in, const int* in_sizes, int n_in,
                              void* d_out, int out_size, void* d_ws, size_t ws_size,
                              hipStream_t stream)
{
    (void)out_size; (void)ws_size;

    // ---- classify flattened pytree inputs by element count ----
    const float* frames = nullptr; const float* maskp = nullptr; const float* wlast = nullptr;
    const float* Wx[3] = {}; const float* Wh[3] = {}; const float* Wm[3] = {};
    const float* Wo[3] = {}; const float* W11[3] = {};
    int nWxBig = 0, nWh = 0, nWm = 0, nWo = 0, n11 = 0;
    for (int i = 0; i < n_in; ++i) {
        const float* p = (const float*)d_in[i];
        switch (in_sizes[i]) {
            case 1572864: frames = p; break;                    // [8,12,32,32,16]
            case 655360:  maskp  = p; break;                    // [8,5,32,32,16]
            case 1024:    wlast  = p; break;                    // [16,64,1,1]
            case 179200:  Wx[0]  = p; break;                    // [448,16,5,5]
            case 716800:  if (nWxBig < 2) Wx[1 + nWxBig++] = p; break;
            case 409600:  if (nWh < 3) Wh[nWh++] = p; break;
            case 307200:  if (nWm < 3) Wm[nWm++] = p; break;
            case 204800:  if (nWo < 3) Wo[nWo++] = p; break;
            case 8192:    if (n11 < 3) W11[n11++] = p; break;
            default: break;
        }
    }

    // ---- workspace bump allocator (256B aligned) ----
    char* wp = (char*)d_ws;
    auto alloc = [&](size_t bytes) -> void* {
        void* p = (void*)wp;
        wp += (bytes + 255) & ~(size_t)255;
        return p;
    };
    const size_t SF = sizeof(float), SH = sizeof(_Float16);

    // recurrent state first (zeroed each call, one memset span)
    char* state0 = wp;
    float*    c_st[3]; for (int i = 0; i < 3; ++i) c_st[i] = (float*)alloc(524288 * SF);
    float*    m_st   = (float*)alloc(524288 * SF);
    _Float16* h16[3]; for (int i = 0; i < 3; ++i) h16[i] = (_Float16*)alloc(524288 * SH);
    _Float16* m16    = (_Float16*)alloc(524288 * SH);
    float*    xgen   = (float*)alloc(131072 * SF);
    size_t stateBytes = (size_t)(wp - state0);

    // packed f16 weights
    const int KxPad[3] = {416, 1600, 1600};      // (16|64)*25 padded to %32
    _Float16* WxP[3]; _Float16* WhP[3]; _Float16* WmP[3]; _Float16* WoP[3]; _Float16* W11P[3];
    for (int i = 0; i < 3; ++i) {
        WxP[i]  = (_Float16*)alloc((size_t)448 * KxPad[i] * SH);
        WhP[i]  = (_Float16*)alloc((size_t)256 * 1600 * SH);
        WmP[i]  = (_Float16*)alloc((size_t)192 * 1600 * SH);
        WoP[i]  = (_Float16*)alloc((size_t)64 * 3200 * SH);
        W11P[i] = (_Float16*)alloc((size_t)64 * 128 * SH);
    }
    _Float16* WlastP = (_Float16*)alloc((size_t)16 * 64 * SH);

    // per-step temporaries
    _Float16* net16 = (_Float16*)alloc(131072 * SH);
    _Float16* mem16 = (_Float16*)alloc(1048576 * SH);
    float* xcc  = (float*)alloc((size_t)3670016 * SF);   // [8,448,1024]
    float* hcc  = (float*)alloc((size_t)2097152 * SF);   // [8,256,1024]
    float* mcc  = (float*)alloc((size_t)1572864 * SF);   // [8,192,1024]
    float* occ  = (float*)alloc(524288 * SF);
    float* opre = (float*)alloc(524288 * SF);
    float* t11  = (float*)alloc(524288 * SF);

    // ---- init: zero recurrent state + loss slot ----
    hipMemsetAsync(state0, 0, stateBytes, stream);
    float* lossp = (float*)d_out + OUT_ELEMS;
    hipMemsetAsync(lossp, 0, sizeof(float), stream);

    // ---- pack weights ----
    auto pack = [&](const float* w, _Float16* o, int Cout, int K, int Kpad) {
        int tot = Cout * Kpad;
        pack_w_kernel<<<(tot + 255) / 256, 256, 0, stream>>>(w, o, Cout, K, Kpad);
    };
    for (int i = 0; i < 3; ++i) {
        int cin = (i == 0) ? FC : HC;
        pack(Wx[i],  WxP[i],  7 * HC, cin * 25, KxPad[i]);
        pack(Wh[i],  WhP[i],  4 * HC, HC * 25, 1600);
        pack(Wm[i],  WmP[i],  3 * HC, HC * 25, 1600);
        pack(Wo[i],  WoP[i],  HC, 2 * HC * 25, 3200);
        pack(W11[i], W11P[i], HC, 2 * HC, 128);
    }
    pack(wlast, WlastP, FC, HC, 64);

    // ---- recurrence ----
    const int EW = 131072 / 256;    // blend/emit grid
    const int GW = 524288 / 256;    // gate/hout grid
    for (int t = 0; t < STEPS; ++t) {
        blend_kernel<<<EW, 256, 0, stream>>>(frames, maskp, xgen, net16, t);
        for (int i = 0; i < 3; ++i) {
            const _Float16* xin = (i == 0) ? net16 : h16[i - 1];
            int cin  = (i == 0) ? FC : HC;
            launch_gemm(stream, WxP[i], xin,    xcc, 7 * HC, cin, KxPad[i], 5);
            layernorm_kernel<<<BATCH, 256, 0, stream>>>(xcc, 7 * HC * HWSZ);
            launch_gemm(stream, WhP[i], h16[i], hcc, 4 * HC, HC, 1600, 5);
            layernorm_kernel<<<BATCH, 256, 0, stream>>>(hcc, 4 * HC * HWSZ);
            launch_gemm(stream, WmP[i], m16,    mcc, 3 * HC, HC, 1600, 5);
            layernorm_kernel<<<BATCH, 256, 0, stream>>>(mcc, 3 * HC * HWSZ);
            gates_kernel<<<GW, 256, 0, stream>>>(xcc, hcc, mcc, c_st[i], m_st,
                                                 opre, mem16, m16);
            launch_gemm(stream, WoP[i],  mem16, occ, HC, 2 * HC, 3200, 5);
            layernorm_kernel<<<BATCH, 256, 0, stream>>>(occ, HC * HWSZ);
            launch_gemm(stream, W11P[i], mem16, t11, HC, 2 * HC, 128, 1);
            hout_kernel<<<GW, 256, 0, stream>>>(opre, occ, t11, h16[i]);
        }
        launch_gemm(stream, WlastP, h16[2], xgen, FC, HC, 64, 1);
        emit_kernel<<<EW, 256, 0, stream>>>(xgen, frames, (float*)d_out, lossp, t);
    }
}
